// Image2BEV_18305150615726
// MI455X (gfx1250) — matrix operand order repaired and checked by reading, not compiled, not run on hardware
//
#include <hip/hip_runtime.h>
#include <hip/hip_bf16.h>

typedef __attribute__((ext_vector_type(16))) _Float16 v16h;
typedef __attribute__((ext_vector_type(8)))  _Float16 v8h;
typedef __attribute__((ext_vector_type(8)))  float    v8f;

#define NVIEW  6
#define DMODEL 256
#define NHEAD  8
#define DHEAD  32
#define NLVL   4
#define NZP    4
#define BEVW   80
#define NQ     (BEVW * BEVW)   /* 6400 */
#define NTOK   7441
#define NTOKP  7456            /* padded to 16 rows */
#define PADH   448.0f
#define PADW   800.0f
#define EPSV   1e-5f

__constant__ int c_lh[4] = {56, 28, 14, 7};
__constant__ int c_lw[4] = {100, 50, 25, 13};
__constant__ int c_ls[4] = {0, 5600, 7000, 7350};

__device__ __forceinline__ int iclampi(int x, int lo, int hi) {
    return x < lo ? lo : (x > hi ? hi : x);
}

__device__ __forceinline__ void lvl_of(int n, int& l, int& nl, int& hw) {
    if (n < 5600)      { l = 0; nl = n;        hw = 5600; }
    else if (n < 7000) { l = 1; nl = n - 5600; hw = 1400; }
    else if (n < 7350) { l = 2; nl = n - 7000; hw = 350;  }
    else               { l = 3; nl = n - 7350; hw = 91;   }
}

// ---------------------------------------------------------------------------
// 1) Reference-point projection: coords[v,q,2], mask[v,q]
// ---------------------------------------------------------------------------
__global__ void proj_kernel(const float* __restrict__ l2i,
                            float* __restrict__ coords,
                            float* __restrict__ mask) {
    int idx = blockIdx.x * blockDim.x + threadIdx.x;
    if (idx >= NVIEW * NQ) return;
    int v = idx / NQ, q = idx % NQ;
    float xs = ((float)(q % BEVW) + 0.5f) / (float)BEVW;
    float ys = ((float)(q / BEVW) + 0.5f) / (float)BEVW;
    float x = xs * 96.0f - 48.0f;
    float y = ys * 96.0f - 48.0f;
    float z = x * 8.0f - 5.0f;   // faithful to ref bug: z from scaled x
    const float* M = l2i + (size_t)v * 16;
    float ip0 = M[0] * x + M[1] * y + M[2]  * z + M[3];
    float ip1 = M[4] * x + M[5] * y + M[6]  * z + M[7];
    float ip2 = M[8] * x + M[9] * y + M[10] * z + M[11];
    float d  = fmaxf(ip2, EPSV);
    float xn = (ip0 / d) / PADW;
    float yn = (ip1 / d) / PADH;
    bool m = (ip2 > EPSV) && (xn > 0.0f) && (xn < 1.0f) && (yn > 0.0f) && (yn < 1.0f);
    coords[(size_t)idx * 2 + 0] = xn;
    coords[(size_t)idx * 2 + 1] = yn;
    mask[idx] = m ? 1.0f : 0.0f;
}

// ---------------------------------------------------------------------------
// Pre-pack kernels: put GEMM operands in f16, fragment-friendly layouts.
// ---------------------------------------------------------------------------
// featT[v][n][k] = (f16) feat_l[v][k][n_local]   (reads coalesced along n)
__global__ void featT_kernel(const float* __restrict__ f0,
                             const float* __restrict__ f1,
                             const float* __restrict__ f2,
                             const float* __restrict__ f3,
                             _Float16* __restrict__ featT) {
    int idx = blockIdx.x * blockDim.x + threadIdx.x;   // (v*32 + kg)*NTOK + n
    if (idx >= NVIEW * 32 * NTOK) return;
    int n = idx % NTOK;
    int t = idx / NTOK;
    int v = t >> 5, kg = t & 31;
    int l, nl, hw;
    lvl_of(n, l, nl, hw);
    const float* f = (l == 0) ? f0 : (l == 1) ? f1 : (l == 2) ? f2 : f3;
    const float* src = f + ((size_t)v * DMODEL + kg * 8) * hw + nl;
    v8h o;
#pragma unroll
    for (int j = 0; j < 8; ++j) o[j] = (_Float16)src[(size_t)j * hw];
    *(v8h*)(featT + ((size_t)v * NTOKP + n) * DMODEL + kg * 8) = o;
}

// Wt[n][k] = (f16) W[k][n]  (column-major f16 weights -> contiguous B frags)
__global__ void packWt_kernel(const float* __restrict__ W,
                              _Float16* __restrict__ Wt, int Ncols) {
    int idx = blockIdx.x * blockDim.x + threadIdx.x;   // n*32 + kg
    if (idx >= Ncols * 32) return;
    int kg = idx & 31;
    int n  = idx >> 5;
    v8h o;
#pragma unroll
    for (int j = 0; j < 8; ++j)
        o[j] = (_Float16)W[(size_t)(kg * 8 + j) * Ncols + n];
    *(v8h*)(Wt + (size_t)n * DMODEL + kg * 8) = o;
}

// plain f32 -> f16 copy
__global__ void packh_kernel(const float* __restrict__ A,
                             _Float16* __restrict__ Ah, int total) {
    int idx = blockIdx.x * blockDim.x + threadIdx.x;
    if (idx < total) Ah[idx] = (_Float16)A[idx];
}

// ---------------------------------------------------------------------------
// 2a) Embedding bias precompute: ebias[v,l,:] = (vemb[v]+lemb[l]) @ Wv + bv
// ---------------------------------------------------------------------------
__global__ void ebias_kernel(const float* __restrict__ vemb,
                             const float* __restrict__ lemb,
                             const float* __restrict__ Wv,
                             const float* __restrict__ bv,
                             float* __restrict__ ebias) {
    int idx = blockIdx.x * blockDim.x + threadIdx.x;   // (v*NLVL+l)*256 + col
    if (idx >= NVIEW * NLVL * DMODEL) return;
    int col = idx & (DMODEL - 1);
    int vl  = idx >> 8;
    int v = vl >> 2, l = vl & 3;
    const float* ve = vemb + v * DMODEL;
    const float* le = lemb + l * DMODEL;
    float s = bv[col];
    for (int k = 0; k < DMODEL; ++k)
        s += (ve[k] + le[k]) * Wv[(size_t)k * DMODEL + col];
    ebias[idx] = s;
}

// ---------------------------------------------------------------------------
// WMMA fragment loaders (f16, contiguous 16B chunks)
//   A frag: K = kb + (hi?8:0) + t (t<8), +16 for t>=8
//   B frag: K = kb + (hi?8:0) + t (t<8), +16 for t>=8
// ---------------------------------------------------------------------------
__device__ __forceinline__ v16h load_afrag(const _Float16* base) {
    v8h a0 = *(const v8h*)(base);
    v8h a1 = *(const v8h*)(base + 16);
    v16h a;
#pragma unroll
    for (int j = 0; j < 8; ++j) { a[j] = a0[j]; a[j + 8] = a1[j]; }
    return a;
}
__device__ __forceinline__ v16h load_bfrag(const _Float16* base) {
    v8h b0 = *(const v8h*)(base);
    v8h b1 = *(const v8h*)(base + 16);
    v16h b;
#pragma unroll
    for (int j = 0; j < 8; ++j) { b[j] = b0[j]; b[j + 8] = b1[j]; }
    return b;
}

// ---------------------------------------------------------------------------
// 2b) Value GEMM: value[v,n,:] = featT[v,n,:] @ WvT + ebias[v,lvl(n),:]
//     One wave per 16x64 tile; A frag reused across 4 column tiles.
// ---------------------------------------------------------------------------
__global__ void value_gemm_kernel(const _Float16* __restrict__ featT,
                                  const _Float16* __restrict__ WvT,
                                  const float* __restrict__ ebias,
                                  float* __restrict__ value) {
    const int TM = NTOKP / 16;     // 466 row tiles
    const int NG = DMODEL / 64;    // 4 groups of 64 cols
    int wave = (blockIdx.x * blockDim.x + threadIdx.x) >> 5;
    int lane = threadIdx.x & 31;
    if (wave >= NVIEW * TM * NG) return;
    int v   = wave / (TM * NG);
    int rem = wave % (TM * NG);
    int tm = rem / NG;
    int colbase = (rem % NG) * 64;
    int low = lane & 15;
    int hi  = lane >> 4;
    int row = tm * 16 + low;

    const _Float16* arow = featT + ((size_t)v * NTOKP + row) * DMODEL;
    v8f acc0 = {}, acc1 = {}, acc2 = {}, acc3 = {};
    for (int kb = 0; kb < DMODEL; kb += 32) {
        v16h a = load_afrag(arow + kb + hi * 8);
#pragma unroll
        for (int g = 0; g < 4; ++g) {
            int col = colbase + g * 16 + low;
            v16h b = load_bfrag(WvT + (size_t)col * DMODEL + kb + hi * 8);
            v8f& acc = (g == 0) ? acc0 : (g == 1) ? acc1 : (g == 2) ? acc2 : acc3;
            acc = __builtin_amdgcn_wmma_f32_16x16x32_f16(false, a, false, b,
                                                         (short)0, acc, false, false);
        }
    }
#pragma unroll
    for (int r = 0; r < 8; ++r) {
        int m = tm * 16 + r + hi * 8;
        if (m >= NTOK) continue;
        int ml, mnl, mhw;
        lvl_of(m, ml, mnl, mhw);
        const float* eb = ebias + ((size_t)v * NLVL + ml) * DMODEL;
        float* vout = value + ((size_t)v * NTOK + m) * DMODEL;
        vout[colbase +  0 + low] = acc0[r] + eb[colbase +  0 + low];
        vout[colbase + 16 + low] = acc1[r] + eb[colbase + 16 + low];
        vout[colbase + 32 + low] = acc2[r] + eb[colbase + 32 + low];
        vout[colbase + 48 + low] = acc3[r] + eb[colbase + 48 + low];
    }
}

// ---------------------------------------------------------------------------
// 3) Generic GEMM: C[M,Ncols] = A_h[M,256] @ Wt_h[Ncols,256]^T + bias (+addend)
//    Requires M % 16 == 0 (true for all call sites, M = 6400).
// ---------------------------------------------------------------------------
__global__ void gemm256h_kernel(const _Float16* __restrict__ A,
                                const _Float16* __restrict__ Wt,
                                const float* __restrict__ bias,
                                const float* __restrict__ addend,
                                float* __restrict__ C,
                                int M, int Ncols) {
    int tilesM = M >> 4;
    int groups = Ncols >> 6;   // 64-col groups
    int wave = (blockIdx.x * blockDim.x + threadIdx.x) >> 5;
    int lane = threadIdx.x & 31;
    if (wave >= tilesM * groups) return;
    int tm = wave / groups;
    int colbase = (wave % groups) * 64;
    int low = lane & 15;
    int hi  = lane >> 4;
    int row = tm * 16 + low;

    const _Float16* arow = A + (size_t)row * DMODEL;
    v8f acc0 = {}, acc1 = {}, acc2 = {}, acc3 = {};
    for (int kb = 0; kb < DMODEL; kb += 32) {
        v16h a = load_afrag(arow + kb + hi * 8);
#pragma unroll
        for (int g = 0; g < 4; ++g) {
            int col = colbase + g * 16 + low;
            v16h b = load_bfrag(Wt + (size_t)col * DMODEL + kb + hi * 8);
            v8f& acc = (g == 0) ? acc0 : (g == 1) ? acc1 : (g == 2) ? acc2 : acc3;
            acc = __builtin_amdgcn_wmma_f32_16x16x32_f16(false, a, false, b,
                                                         (short)0, acc, false, false);
        }
    }
#pragma unroll
    for (int r = 0; r < 8; ++r) {
        int m = tm * 16 + r + hi * 8;
#pragma unroll
        for (int g = 0; g < 4; ++g) {
            int col = colbase + g * 16 + low;
            const v8f& acc = (g == 0) ? acc0 : (g == 1) ? acc1 : (g == 2) ? acc2 : acc3;
            float o = acc[r] + bias[col];
            if (addend) o += addend[(size_t)m * Ncols + col];
            C[(size_t)m * Ncols + col] = o;
        }
    }
}

// ---------------------------------------------------------------------------
// 4) Softmax over the 16 (L*ZP) logits per (q, head); in-place.
// ---------------------------------------------------------------------------
__global__ void softmax16_kernel(float* __restrict__ aw) {
    int idx = blockIdx.x * blockDim.x + threadIdx.x;   // q*NHEAD + h
    if (idx >= NQ * NHEAD) return;
    float* p = aw + (size_t)idx * 16;
    float mx = -1e30f;
#pragma unroll
    for (int i = 0; i < 16; ++i) mx = fmaxf(mx, p[i]);
    float e[16];
    float s = 0.0f;
#pragma unroll
    for (int i = 0; i < 16; ++i) { e[i] = __expf(p[i] - mx); s += e[i]; }
    float inv = 1.0f / s;
#pragma unroll
    for (int i = 0; i < 16; ++i) p[i] = e[i] * inv;
}

// ---------------------------------------------------------------------------
// 5) Deformable bilinear sampling: one wave per (v, q, head); lane = dh.
// ---------------------------------------------------------------------------
__global__ void sample_kernel(const float* __restrict__ value,
                              const float* __restrict__ coords,
                              const float* __restrict__ aw,
                              const float* __restrict__ off,
                              float* __restrict__ attn) {
    int wave = (blockIdx.x * blockDim.x + threadIdx.x) >> 5;
    int lane = threadIdx.x & 31;   // dh
    if (wave >= NVIEW * NQ * NHEAD) return;
    int v   = wave / (NQ * NHEAD);
    int rem = wave % (NQ * NHEAD);
    int q = rem / NHEAD, h = rem % NHEAD;

    float xn = coords[((size_t)v * NQ + q) * 2 + 0];
    float yn = coords[((size_t)v * NQ + q) * 2 + 1];
    const float* offp = off + (size_t)q * 512 + h * 32;   // [l*8 + zp*2 + {x,y}]
    const float* awp  = aw  + (size_t)q * 256 + h * 16;   // [l*4 + zp]
    const float* vb   = value + (size_t)v * NTOK * DMODEL + h * DHEAD + lane;

    float acc = 0.0f;
#pragma unroll
    for (int l = 0; l < NLVL; ++l) {
        int hh = c_lh[l], ww = c_lw[l], st = c_ls[l];
#pragma unroll
        for (int zp = 0; zp < NZP; ++zp) {
            float gx = xn * (float)ww + offp[l * 8 + zp * 2 + 0] - 0.5f;
            float gy = yn * (float)hh + offp[l * 8 + zp * 2 + 1] - 0.5f;
            float fx = floorf(gx), fy = floorf(gy);
            int x0 = (int)fx, y0 = (int)fy;
            float wx1 = gx - fx, wy1 = gy - fy;
            float wx0 = 1.0f - wx1, wy0 = 1.0f - wy1;
            float wgt = awp[l * 4 + zp];

            float s = 0.0f;
#pragma unroll
            for (int c = 0; c < 4; ++c) {
                int xi = x0 + (c & 1);
                int yi = y0 + (c >> 1);
                float cw = ((c & 1) ? wx1 : wx0) * ((c >> 1) ? wy1 : wy0);
                bool valid = (xi >= 0) && (xi < ww) && (yi >= 0) && (yi < hh);
                int lin = iclampi(yi, 0, hh - 1) * ww + iclampi(xi, 0, ww - 1);
                float g = vb[(size_t)(st + lin) * DMODEL];
                s += valid ? g * cw : 0.0f;
            }
            acc += s * wgt;
        }
    }
    attn[((size_t)v * NQ + q) * DMODEL + h * DHEAD + lane] = acc;
}

// ---------------------------------------------------------------------------
// 6) Masked mean over views -> slots_h[q, d] (f16, feeds final WMMA GEMM)
// ---------------------------------------------------------------------------
__global__ void reduce_kernel(const float* __restrict__ attn,
                              const float* __restrict__ mask,
                              _Float16* __restrict__ slots_h) {
    int idx = blockIdx.x * blockDim.x + threadIdx.x;   // q*DMODEL + d
    if (idx >= NQ * DMODEL) return;
    int q = idx / DMODEL;
    float s = 0.0f, cnt = 0.0f;
#pragma unroll
    for (int v = 0; v < NVIEW; ++v) {
        float m = mask[v * NQ + q];
        cnt += m;
        s += attn[(size_t)v * NQ * DMODEL + idx] * m;
    }
    slots_h[idx] = (_Float16)(s / fmaxf(cnt, 1.0f));
}

// ---------------------------------------------------------------------------
extern "C" void kernel_launch(void* const* d_in, const int* in_sizes, int n_in,
                              void* d_out, int out_size, void* d_ws, size_t ws_size,
                              hipStream_t stream) {
    (void)in_sizes; (void)n_in; (void)out_size; (void)ws_size;
    const float* f0   = (const float*)d_in[0];
    const float* f1   = (const float*)d_in[1];
    const float* f2   = (const float*)d_in[2];
    const float* f3   = (const float*)d_in[3];
    const float* l2i  = (const float*)d_in[4];
    const float* bev  = (const float*)d_in[5];
    const float* lemb = (const float*)d_in[6];
    const float* vemb = (const float*)d_in[7];
    const float* Wv   = (const float*)d_in[8];
    const float* bv   = (const float*)d_in[9];
    const float* Woff = (const float*)d_in[10];
    const float* boff = (const float*)d_in[11];
    const float* Waw  = (const float*)d_in[12];
    const float* baw  = (const float*)d_in[13];
    const float* Wout = (const float*)d_in[14];
    const float* bout = (const float*)d_in[15];
    float* out = (float*)d_out;

    // bump allocator on d_ws (256B aligned chunks)
    char* ws = (char*)d_ws;
    auto take = [&](size_t bytes) -> void* {
        void* p = (void*)ws;
        ws += (bytes + 255) & ~(size_t)255;
        return p;
    };
    float*    coords  = (float*)take((size_t)NVIEW * NQ * 2 * 4);
    float*    maskb   = (float*)take((size_t)NVIEW * NQ * 4);
    float*    ebias   = (float*)take((size_t)NVIEW * NLVL * DMODEL * 4);
    _Float16* featT   = (_Float16*)take((size_t)NVIEW * NTOKP * DMODEL * 2);
    _Float16* WvT     = (_Float16*)take((size_t)DMODEL * DMODEL * 2);
    _Float16* WoffT   = (_Float16*)take((size_t)512 * DMODEL * 2);
    _Float16* WawT    = (_Float16*)take((size_t)DMODEL * DMODEL * 2);
    _Float16* WoutT   = (_Float16*)take((size_t)DMODEL * DMODEL * 2);
    _Float16* bev_h   = (_Float16*)take((size_t)NQ * DMODEL * 2);
    _Float16* slots_h = (_Float16*)take((size_t)NQ * DMODEL * 2);
    float*    value   = (float*)take((size_t)NVIEW * NTOK * DMODEL * 4);
    float*    offb    = (float*)take((size_t)NQ * 512 * 4);
    float*    awb     = (float*)take((size_t)NQ * 256 * 4);
    float*    attn    = (float*)take((size_t)NVIEW * NQ * DMODEL * 4);

    // 1) projection
    proj_kernel<<<(NVIEW * NQ + 255) / 256, 256, 0, stream>>>(l2i, coords, maskb);

    // pre-pack operands to f16 fragment-friendly layouts
    {
        int tot = NVIEW * 32 * NTOK;
        featT_kernel<<<(tot + 255) / 256, 256, 0, stream>>>(f0, f1, f2, f3, featT);
    }
    packWt_kernel<<<(DMODEL * 32 + 255) / 256, 256, 0, stream>>>(Wv,   WvT,   DMODEL);
    packWt_kernel<<<(512    * 32 + 255) / 256, 256, 0, stream>>>(Woff, WoffT, 512);
    packWt_kernel<<<(DMODEL * 32 + 255) / 256, 256, 0, stream>>>(Waw,  WawT,  DMODEL);
    packWt_kernel<<<(DMODEL * 32 + 255) / 256, 256, 0, stream>>>(Wout, WoutT, DMODEL);
    packh_kernel<<<(NQ * DMODEL + 255) / 256, 256, 0, stream>>>(bev, bev_h, NQ * DMODEL);

    // 2a) embedding bias precompute
    ebias_kernel<<<(NVIEW * NLVL * DMODEL + 255) / 256, 256, 0, stream>>>(
        vemb, lemb, Wv, bv, ebias);

    // 2b) value projection GEMM
    {
        int waves = NVIEW * (NTOKP / 16) * (DMODEL / 64);
        value_gemm_kernel<<<(waves * 32 + 255) / 256, 256, 0, stream>>>(
            featT, WvT, ebias, value);
    }
    // 3) offsets GEMM  (M=6400, N=512)
    {
        int waves = (NQ / 16) * (512 / 64);
        gemm256h_kernel<<<(waves * 32 + 255) / 256, 256, 0, stream>>>(
            bev_h, WoffT, boff, nullptr, offb, NQ, 512);
    }
    // 3b) attention-weight GEMM (M=6400, N=256) + softmax
    {
        int waves = (NQ / 16) * (256 / 64);
        gemm256h_kernel<<<(waves * 32 + 255) / 256, 256, 0, stream>>>(
            bev_h, WawT, baw, nullptr, awb, NQ, 256);
    }
    softmax16_kernel<<<(NQ * NHEAD + 255) / 256, 256, 0, stream>>>(awb);

    // 5) deformable bilinear sampling
    {
        int waves = NVIEW * NQ * NHEAD;
        sample_kernel<<<(waves * 32 + 255) / 256, 256, 0, stream>>>(
            value, coords, awb, offb, attn);
    }
    // 6) masked view mean (emits f16 A for the final GEMM)
    reduce_kernel<<<(NQ * DMODEL + 255) / 256, 256, 0, stream>>>(attn, maskb, slots_h);

    // 7) output projection + residual, straight into d_out
    {
        int waves = (NQ / 16) * (256 / 64);
        gemm256h_kernel<<<(waves * 32 + 255) / 256, 256, 0, stream>>>(
            slots_h, WoutT, bout, bev, out, NQ, 256);
    }
}
